// Head_22917945491929
// MI455X (gfx1250) — matrix operand order repaired and checked by reading, not compiled, not run on hardware
//
#include <hip/hip_runtime.h>
#include <hip/hip_bf16.h>

#define B_    8
#define T_    2048
#define EMB_  1024
#define HEAD_ 1024

typedef __attribute__((ext_vector_type(16))) __bf16 v16bf;
typedef __attribute__((ext_vector_type(8)))  float  v8f;

typedef unsigned int u32x4 __attribute__((ext_vector_type(4)));
typedef int          i32x8 __attribute__((ext_vector_type(8)));
typedef int          i32x4 __attribute__((ext_vector_type(4)));

union FragBF { v16bf v; unsigned u[8]; };

// f32 pair -> packed bf16 dword (compiler fuses into v_cvt_pk_bf16_f32)
__device__ inline unsigned pack2_bf16(float a, float b) {
    unsigned lo = (unsigned)__bfloat16_as_ushort(__float2bfloat16(a));
    unsigned hi = (unsigned)__bfloat16_as_ushort(__float2bfloat16(b));
    return lo | (hi << 16);
}

// ---------------------------------------------------------------------------
// Projection: Q = x Wq^T, K = x Wk^T, V^T (stored [B, HEAD, T]) = (x Wv^T)^T
// grid: (B*T/16, HEAD/128, 3), block: 256 (8 waves), one 16x16 tile per wave.
// ---------------------------------------------------------------------------
__global__ __launch_bounds__(256) void proj_kernel(
    const float* __restrict__ x,  const float* __restrict__ Wq,
    const float* __restrict__ Wk, const float* __restrict__ Wv,
    __hip_bfloat16* __restrict__ Q, __hip_bfloat16* __restrict__ K,
    __hip_bfloat16* __restrict__ Vt)
{
    __shared__ __hip_bfloat16 sX[16 * EMB_];   // 32 KB, 16 x 1024 bf16 row-major

    const int tid  = threadIdx.x;
    const int wave = tid >> 5;
    const int lane = tid & 31;
    const int hlf  = lane >> 4;
    const int n    = lane & 15;

    const int row0  = blockIdx.x * 16;               // row in [0, B*T)
    const int hbase = blockIdx.y * 128 + wave * 16;  // output column base
    const int which = blockIdx.z;                    // 0=Q 1=K 2=V

    const float* W = (which == 0) ? Wq : (which == 1) ? Wk : Wv;

    // stage x tile (16 x EMB) into LDS as bf16 (rows are contiguous in memory)
    {
        const float* xt = x + (size_t)row0 * EMB_;
        for (int i = tid * 4; i < 16 * EMB_; i += 256 * 4) {
            float4 f = *(const float4*)(xt + i);
            unsigned* dst = (unsigned*)(sX + i);
            dst[0] = pack2_bf16(f.x, f.y);
            dst[1] = pack2_bf16(f.z, f.w);
        }
    }
    __syncthreads();

    v8f acc = {};
    for (int k0 = 0; k0 < EMB_; k0 += 32) {
        FragBF fa, fb;
        // A fragment: rows of x tile   (m = lane&15)
        #pragma unroll
        for (int r = 0; r < 8; ++r) {
            int k = (hlf << 4) + (r << 1) + k0;
            fa.u[r] = *(const unsigned*)(sX + n * EMB_ + k);
        }
        // B fragment: B[k][n] = W[hbase+n][k]   (f32 -> bf16 on the fly)
        const float* wrow = W + (size_t)(hbase + n) * EMB_;
        #pragma unroll
        for (int r = 0; r < 8; ++r) {
            int k = (hlf << 4) + (r << 1) + k0;
            float2 f = *(const float2*)(wrow + k);
            fb.u[r] = pack2_bf16(f.x, f.y);
        }
        acc = __builtin_amdgcn_wmma_f32_16x16x32_bf16(false, fa.v, false, fb.v,
                                                      (short)0, acc, false, false);
    }

    if (which < 2) {
        __hip_bfloat16* out = (which == 0) ? Q : K;
        #pragma unroll
        for (int v = 0; v < 8; ++v) {
            int m = (hlf << 3) + v;
            out[(size_t)(row0 + m) * HEAD_ + hbase + n] = __float2bfloat16(acc[v]);
        }
    } else {
        int bb = row0 / T_;
        int t0 = row0 % T_;
        #pragma unroll
        for (int v = 0; v < 8; ++v) {
            int m = (hlf << 3) + v;
            Vt[(size_t)bb * HEAD_ * T_ + (size_t)(hbase + n) * T_ + (t0 + m)] =
                __float2bfloat16(acc[v]);
        }
    }
}

// ---------------------------------------------------------------------------
// Flash attention: workgroup (8 waves) owns 16 query rows; iterates over
// 128-wide causal K/V blocks. grid: (T/16, B), block: 256.
// Q tile staged into LDS by the Tensor Data Mover (one TDM descriptor).
// ---------------------------------------------------------------------------
__global__ __launch_bounds__(256) void attn_kernel(
    const __hip_bfloat16* __restrict__ Q,
    const __hip_bfloat16* __restrict__ K,
    const __hip_bfloat16* __restrict__ Vt,
    float* __restrict__ out)
{
    __shared__ __hip_bfloat16 sQ[16 * HEAD_];  // 32 KB
    __shared__ __hip_bfloat16 sP[16 * 128];    // 4 KB
    __shared__ float sRedM[8][16];
    __shared__ float sRedS[8][16];

    const int tid  = threadIdx.x;
    const int wave = tid >> 5;
    const int lane = tid & 31;
    const int hlf  = lane >> 4;
    const int n    = lane & 15;

    const int i0 = blockIdx.x * 16;   // query row tile base
    const int b  = blockIdx.y;

    const __hip_bfloat16* Qb = Q  + (size_t)b * T_ * HEAD_;
    const __hip_bfloat16* Kb = K  + (size_t)b * T_ * HEAD_;
    const __hip_bfloat16* Vb = Vt + (size_t)b * HEAD_ * T_;

    // ---- TDM: DMA the 16x1024 bf16 Q tile into LDS (wave 0 issues) ----
    if (wave == 0) {
        unsigned lds_addr = (unsigned)(size_t)(void*)sQ;  // LDS byte offset
        unsigned long long ga =
            (unsigned long long)(const void*)(Qb + (size_t)i0 * HEAD_);
        // D# group0: count=1 | lds_addr | global_addr[56:0] | type=2
        u32x4 g0 = { 1u,
                     lds_addr,
                     (unsigned)(ga & 0xFFFFFFFFu),
                     (unsigned)((ga >> 32) & 0x01FFFFFFu) | (2u << 30) };
        // D# group1: data_size=2B; tensor 1024 x 16, stride 1024; tile 1024 x 16
        i32x8 g1 = { (int)0x00010000,        // wg_mask=0, data_size=1 (2 bytes)
                     (int)(1024u << 16),     // tensor_dim0[15:0]  (bits 63:48)
                     (int)(16u   << 16),     // tensor_dim1[15:0]  (bits 95:80)
                     (int)(1024u << 16),     // tile_dim0          (bits 127:112)
                     16,                     // tile_dim1          (bits 143:128)
                     1024,                   // tensor_dim0_stride[31:0]
                     0, 0 };
        i32x4 gz4 = { 0, 0, 0, 0 };
        i32x8 gz8 = { 0, 0, 0, 0, 0, 0, 0, 0 };
        __builtin_amdgcn_tensor_load_to_lds(g0, g1, gz4, gz4, gz8, 0);
        __builtin_amdgcn_s_wait_tensorcnt(0);
    }
    __syncthreads();

    const float scale = 0.03125f;     // EMB^-0.5 = 1/32 (reference scales by C)
    const float NEGINF = -__builtin_inff();

    float m_run[8], l_run[8];
    v8f o_acc[8];
    v8f vzero = {};
    #pragma unroll
    for (int t = 0; t < 8; ++t) o_acc[t] = vzero;
    #pragma unroll
    for (int v = 0; v < 8; ++v) { m_run[v] = -3.0e38f; l_run[v] = 0.0f; }

    const int nblocks = i0 / 128 + 1;
    for (int blk = 0; blk < nblocks; ++blk) {
        const int j0 = blk * 128;
        const int jt = j0 + wave * 16;   // this wave's S-tile column base

        // prefetch next block's K rows / V rows into cache (global_prefetch_b8)
        if (blk + 1 < nblocks) {
            __builtin_prefetch(Kb + (size_t)(jt + 128 + n) * HEAD_, 0, 3);
            __builtin_prefetch(Vb + (size_t)(wave * 128 + n) * T_ + j0 + 128, 0, 3);
        }

        // ---- S tile = Q * K^T over head dim ----
        v8f s = vzero;
        for (int k0 = 0; k0 < HEAD_; k0 += 32) {
            FragBF fa, fb;
            #pragma unroll
            for (int r = 0; r < 8; ++r) {
                int k = (hlf << 4) + (r << 1) + k0;
                fa.u[r] = *(const unsigned*)(sQ + n * HEAD_ + k);
            }
            const __hip_bfloat16* krow = Kb + (size_t)(jt + n) * HEAD_;
            #pragma unroll
            for (int r = 0; r < 8; ++r) {
                int k = (hlf << 4) + (r << 1) + k0;
                fb.u[r] = *(const unsigned*)(krow + k);
            }
            s = __builtin_amdgcn_wmma_f32_16x16x32_bf16(false, fa.v, false, fb.v,
                                                        (short)0, s, false, false);
        }

        // scale + causal mask (only blocks crossing the diagonal)
        const bool need_mask = (j0 + 128 > i0);
        #pragma unroll
        for (int v = 0; v < 8; ++v) {
            float sv = s[v] * scale;
            if (need_mask) {
                int ig = i0 + (hlf << 3) + v;
                int jg = jt + n;
                if (jg > ig) sv = NEGINF;
            }
            s[v] = sv;
        }

        // in-wave row max over the 16 lanes sharing `hlf`
        float bmax[8];
        #pragma unroll
        for (int v = 0; v < 8; ++v) {
            float xv = s[v];
            xv = fmaxf(xv, __shfl_xor(xv, 1, 32));
            xv = fmaxf(xv, __shfl_xor(xv, 2, 32));
            xv = fmaxf(xv, __shfl_xor(xv, 4, 32));
            xv = fmaxf(xv, __shfl_xor(xv, 8, 32));
            bmax[v] = xv;
        }

        __syncthreads();                       // sP/sRed reuse from prev iter
        if (n == 0) {
            #pragma unroll
            for (int v = 0; v < 8; ++v) sRedM[wave][(hlf << 3) + v] = bmax[v];
        }
        __syncthreads();

        // combined block max -> new running max, correction factor
        float alpha[8], p[8];
        #pragma unroll
        for (int v = 0; v < 8; ++v) {
            int row = (hlf << 3) + v;
            float bm = sRedM[0][row];
            #pragma unroll
            for (int w = 1; w < 8; ++w) bm = fmaxf(bm, sRedM[w][row]);
            float mnew = fmaxf(m_run[v], bm);
            alpha[v] = __expf(m_run[v] - mnew);
            m_run[v] = mnew;
            p[v] = __expf(s[v] - mnew);
        }

        // in-wave row sums of P
        #pragma unroll
        for (int v = 0; v < 8; ++v) {
            float xv = p[v];
            xv += __shfl_xor(xv, 1, 32);
            xv += __shfl_xor(xv, 2, 32);
            xv += __shfl_xor(xv, 4, 32);
            xv += __shfl_xor(xv, 8, 32);
            if (n == 0) sRedS[wave][(hlf << 3) + v] = xv;
        }

        // write P tile (bf16) to LDS
        #pragma unroll
        for (int v = 0; v < 8; ++v)
            sP[((hlf << 3) + v) * 128 + wave * 16 + n] = __float2bfloat16(p[v]);
        __syncthreads();

        // update l, rescale O accumulators
        #pragma unroll
        for (int v = 0; v < 8; ++v) {
            int row = (hlf << 3) + v;
            float tot = 0.0f;
            #pragma unroll
            for (int w = 0; w < 8; ++w) tot += sRedS[w][row];
            l_run[v] = l_run[v] * alpha[v] + tot;
        }
        #pragma unroll
        for (int t = 0; t < 8; ++t)
            #pragma unroll
            for (int v = 0; v < 8; ++v) o_acc[t][v] *= alpha[v];

        // ---- O += P * V : each wave owns head columns [wave*128, wave*128+128) ----
        #pragma unroll
        for (int k0 = 0; k0 < 128; k0 += 32) {
            FragBF fp;
            #pragma unroll
            for (int r = 0; r < 8; ++r) {
                int k = (hlf << 4) + (r << 1) + k0;
                fp.u[r] = *(const unsigned*)(sP + n * 128 + k);
            }
            #pragma unroll
            for (int t = 0; t < 8; ++t) {
                FragBF fv;
                const __hip_bfloat16* vrow =
                    Vb + (size_t)(wave * 128 + t * 16 + n) * T_ + j0;
                #pragma unroll
                for (int r = 0; r < 8; ++r) {
                    int k = (hlf << 4) + (r << 1) + k0;
                    fv.u[r] = *(const unsigned*)(vrow + k);
                }
                o_acc[t] = __builtin_amdgcn_wmma_f32_16x16x32_bf16(
                    false, fp.v, false, fv.v, (short)0, o_acc[t], false, false);
            }
        }
    }

    // final normalize + store f32
    #pragma unroll
    for (int t = 0; t < 8; ++t) {
        int h = wave * 128 + t * 16 + n;
        #pragma unroll
        for (int v = 0; v < 8; ++v) {
            int m = (hlf << 3) + v;
            out[((size_t)b * T_ + i0 + m) * HEAD_ + h] = o_acc[t][v] / l_run[v];
        }
    }
}

extern "C" void kernel_launch(void* const* d_in, const int* in_sizes, int n_in,
                              void* d_out, int out_size, void* d_ws, size_t ws_size,
                              hipStream_t stream)
{
    const float* x  = (const float*)d_in[0];
    const float* Wq = (const float*)d_in[1];
    const float* Wk = (const float*)d_in[2];
    const float* Wv = (const float*)d_in[3];
    float* out = (float*)d_out;

    const size_t nqkv = (size_t)B_ * T_ * HEAD_;
    if (ws_size < 3 * nqkv * sizeof(__hip_bfloat16)) return;

    __hip_bfloat16* Q  = (__hip_bfloat16*)d_ws;
    __hip_bfloat16* K  = Q + nqkv;
    __hip_bfloat16* Vt = K + nqkv;

    dim3 gridP((B_ * T_) / 16, HEAD_ / 128, 3);
    proj_kernel<<<gridP, 256, 0, stream>>>(x, Wq, Wk, Wv, Q, K, Vt);

    dim3 gridA(T_ / 16, B_);
    attn_kernel<<<gridA, 256, 0, stream>>>(Q, K, Vt, out);
}
